// EdgeFeatureAttentionGNN_34230889349206
// MI455X (gfx1250) — matrix operand order repaired and checked by reading, not compiled, not run on hardware
//
#include <hip/hip_runtime.h>
#include <hip/hip_bf16.h>
#include <math.h>

// ---------------------------------------------------------------------------
// GAT with edge features, MI455X (gfx1250, wave32).
//   h   = node_feats @ W            -> bf16 WMMA GEMM (v_wmma_f32_16x16x32_bf16)
//   ee  = edge_feats @ fold(We,ae)  -> tiny GEMV fused into edge pass (fe never materialized)
//   el/er reductions, segment softmax via ordered-uint atomicMax + float atomicAdd
//   out = segment_sum(alpha * h[src]) via float atomics (h & out fit in 192MB L2)
// ---------------------------------------------------------------------------

typedef __attribute__((ext_vector_type(16))) __bf16 v16bf;
typedef __attribute__((ext_vector_type(8)))  float  v8f;

#define NODE_IN 256
#define EDGE_IN 64
#define HEADS   4
#define DOUT    64
#define HD      (HEADS * DOUT)   // 256

// Packed bf16 pair in one dword: lo in [15:0], hi in [31:16] (RNE via __bf16 cast)
__device__ __forceinline__ unsigned pack_bf16(float lo, float hi) {
    unsigned short l = __builtin_bit_cast(unsigned short, (__bf16)lo);
    unsigned short h = __builtin_bit_cast(unsigned short, (__bf16)hi);
    return (unsigned)l | ((unsigned)h << 16);
}

// 8-dword fragment <-> v16bf view
union FragU {
    unsigned u[8];
    uint4    q[2];
    v16bf    v;
};

// ---- float <-> order-preserving uint (for atomic float max) ----
__device__ __forceinline__ unsigned f2o(float f) {
    unsigned u = __float_as_uint(f);
    return (u & 0x80000000u) ? ~u : (u | 0x80000000u);
}
__device__ __forceinline__ float o2f(unsigned u) {
    return (u & 0x80000000u) ? __uint_as_float(u & 0x7FFFFFFFu)
                             : __uint_as_float(~u);
}

// ---------------------------------------------------------------------------
// 0) Zero-init: out, denom, emax (sentinel 0u == "empty segment")
// ---------------------------------------------------------------------------
__global__ void k_init(float* __restrict__ out, float* __restrict__ denom,
                       unsigned* __restrict__ emaxu, int n_out, int n_nh) {
    int t = blockIdx.x * blockDim.x + threadIdx.x;
    int stride = gridDim.x * blockDim.x;
    for (int i = t; i < n_out; i += stride) out[i] = 0.0f;
    for (int i = t; i < n_nh; i += stride) { denom[i] = 0.0f; emaxu[i] = 0u; }
}

// ---------------------------------------------------------------------------
// 1) Fold We with attn_e:  wered[k][h] = sum_d We[k][h*64+d] * attn_e[h][d]
// ---------------------------------------------------------------------------
__global__ void k_wered(const float* __restrict__ We, const float* __restrict__ attn_e,
                        float* __restrict__ wered) {
    int t = threadIdx.x;
    int k = t >> 2, hh = t & 3;
    const float* wr = We + (size_t)k * HD + hh * DOUT;
    const float* ar = attn_e + hh * DOUT;
    float s = 0.0f;
    #pragma unroll 8
    for (int d = 0; d < DOUT; ++d) s += wr[d] * ar[d];
    wered[k * HEADS + hh] = s;
}

// ---------------------------------------------------------------------------
// 2) h = node_feats @ W via bf16 WMMA.
//    Block = 256 thr (8 waves); waves tiled 2(M) x 4(N); wave does 16x64.
//    LDS holds the 32-K W slice COLUMN-MAJOR as packed bf16 pairs, double
//    buffered (one barrier per K-step):
//      Wl[buf][col][k2] = { W[ks+2*k2][col], W[ks+2*k2+1][col] }
//    B fragment (col fixed, K 8h..8h+7 & 16+8h..+7, as A) = dwords 4h..4h+3 & 4h+8..4h+11 ->
//    two ds_load_b128; stride 20 dwords => conflict-free read phases.
// ---------------------------------------------------------------------------
__global__ __launch_bounds__(256) void k_gemm(const float* __restrict__ A,
                                              const float* __restrict__ Wt,
                                              float* __restrict__ Hout, int M) {
    __shared__ unsigned Wl[2][HD][20];    // 2 x 256 cols x (16+4 pad) dwords = 40KB

    const int tid  = threadIdx.x;
    const int wave = tid >> 5;
    const int lane = tid & 31;
    const int l16  = lane & 15;
    const int hi   = lane >> 4;           // K-half selector
    const int mw   = wave >> 2;           // 0..1
    const int nw   = wave & 3;            // 0..3
    const int tileM = blockIdx.x * 32;

    v8f acc[4] = {};

    const int rowA = min(tileM + mw * 16 + l16, M - 1);
    const float* Arow = A + (size_t)rowA * NODE_IN;

    for (int ks = 0; ks < NODE_IN; ks += 32) {
        const int buf = (ks >> 5) & 1;

        // ---- stage W[ks..ks+32) x [0..256) column-major packed bf16 ----
        // thread t owns column t; row-wise reads are wave-coalesced.
        #pragma unroll
        for (int k2 = 0; k2 < 16; ++k2) {
            float w0 = Wt[(size_t)(ks + 2 * k2)     * HD + tid];
            float w1 = Wt[(size_t)(ks + 2 * k2 + 1) * HD + tid];
            Wl[buf][tid][k2] = pack_bf16(w0, w1);
        }
        __syncthreads();

        // ---- A fragment (16x32 bf16): lane l16 = row; hi picks K sub-blocks
        //      (lanes 0-15: K 0-7 & 16-23; lanes 16-31: K 8-15 & 24-31) ----
        float4 a0 = *(const float4*)(Arow + ks + hi * 8);
        float4 a1 = *(const float4*)(Arow + ks + hi * 8 + 4);
        float4 a2 = *(const float4*)(Arow + ks + 16 + hi * 8);
        float4 a3 = *(const float4*)(Arow + ks + 16 + hi * 8 + 4);
        FragU au;
        au.u[0] = pack_bf16(a0.x, a0.y); au.u[1] = pack_bf16(a0.z, a0.w);
        au.u[2] = pack_bf16(a1.x, a1.y); au.u[3] = pack_bf16(a1.z, a1.w);
        au.u[4] = pack_bf16(a2.x, a2.y); au.u[5] = pack_bf16(a2.z, a2.w);
        au.u[6] = pack_bf16(a3.x, a3.y); au.u[7] = pack_bf16(a3.z, a3.w);
        const v16bf a = au.v;

        // ---- pre-load all 4 B fragments (8x ds_load_b128), then WMMA x4 ----
        FragU bu[4];
        #pragma unroll
        for (int t = 0; t < 4; ++t) {
            const unsigned* bp = &Wl[buf][nw * 64 + t * 16 + l16][0] + hi * 4;
            bu[t].q[0] = *(const uint4*)(bp);
            bu[t].q[1] = *(const uint4*)(bp + 8);
        }
        #pragma unroll
        for (int t = 0; t < 4; ++t) {
            acc[t] = __builtin_amdgcn_wmma_f32_16x16x32_bf16(
                false, a, false, bu[t].v, (short)0, acc[t], false, false);
        }
        // no trailing barrier: next iteration stages the other LDS buffer,
        // and its barrier orders re-use of this one.
    }

    // ---- store D (f32 16x16: VGPR r -> M = hi*8 + r, N = l16) ----
    // Fast uniform path for full tiles: one base pointer + immediate offsets.
    float* pbase = Hout + (size_t)(tileM + mw * 16 + hi * 8) * HD + nw * 64 + l16;
    if (tileM + 32 <= M) {
        #pragma unroll
        for (int t = 0; t < 4; ++t)
            #pragma unroll
            for (int r = 0; r < 8; ++r)
                pbase[r * HD + t * 16] = acc[t][r];
    } else {
        #pragma unroll
        for (int t = 0; t < 4; ++t)
            #pragma unroll
            for (int r = 0; r < 8; ++r) {
                int row = tileM + mw * 16 + hi * 8 + r;
                if (row < M) pbase[r * HD + t * 16] = acc[t][r];
            }
    }
}

// ---------------------------------------------------------------------------
// 3) el/er node reductions: thread t -> (n = t>>2, h = t&3)
// ---------------------------------------------------------------------------
__global__ void k_lr(const float* __restrict__ hfeat, const float* __restrict__ attn_l,
                     const float* __restrict__ attn_r, float* __restrict__ el,
                     float* __restrict__ er, int N) {
    int t = blockIdx.x * blockDim.x + threadIdx.x;
    if (t >= N * HEADS) return;
    int n = t >> 2, hh = t & 3;
    const float* hr = hfeat + (size_t)n * HD + hh * DOUT;
    const float* al = attn_l + hh * DOUT;
    const float* ar = attn_r + hh * DOUT;
    float sl = 0.0f, sr = 0.0f;
    #pragma unroll 4
    for (int d = 0; d < DOUT; d += 4) {
        float4 x = *(const float4*)(hr + d);
        sl += x.x * al[d] + x.y * al[d + 1] + x.z * al[d + 2] + x.w * al[d + 3];
        sr += x.x * ar[d] + x.y * ar[d + 1] + x.z * ar[d + 2] + x.w * ar[d + 3];
    }
    el[t] = sl; er[t] = sr;
}

// ---------------------------------------------------------------------------
// 4) Edge pass 1: ee (folded GEMV) + leaky_relu(el[src]+er[dst]+ee),
//    store e, atomic segment-max (ordered uint).
// ---------------------------------------------------------------------------
__global__ void k_edge1(const float* __restrict__ ef, const float* __restrict__ wered,
                        const float* __restrict__ el, const float* __restrict__ er,
                        const int* __restrict__ src, const int* __restrict__ dst,
                        float* __restrict__ ebuf, unsigned* __restrict__ emaxu, int E) {
    int e = blockIdx.x * blockDim.x + threadIdx.x;
    if (e >= E) return;
    const float* row = ef + (size_t)e * EDGE_IN;
    float s0 = 0.f, s1 = 0.f, s2 = 0.f, s3 = 0.f;
    #pragma unroll 4
    for (int k = 0; k < EDGE_IN; k += 4) {
        float4 x = *(const float4*)(row + k);
        const float* w = wered + k * HEADS;      // uniform across wave -> scalar loads
        s0 += x.x * w[0] + x.y * w[4] + x.z * w[8]  + x.w * w[12];
        s1 += x.x * w[1] + x.y * w[5] + x.z * w[9]  + x.w * w[13];
        s2 += x.x * w[2] + x.y * w[6] + x.z * w[10] + x.w * w[14];
        s3 += x.x * w[3] + x.y * w[7] + x.z * w[11] + x.w * w[15];
    }
    int s = src[e], d = dst[e];
    float ev[4] = { el[s * 4 + 0] + er[d * 4 + 0] + s0,
                    el[s * 4 + 1] + er[d * 4 + 1] + s1,
                    el[s * 4 + 2] + er[d * 4 + 2] + s2,
                    el[s * 4 + 3] + er[d * 4 + 3] + s3 };
    float4 st;
    #pragma unroll
    for (int hh = 0; hh < 4; ++hh) {
        float v = ev[hh];
        v = v > 0.0f ? v : 0.2f * v;             // leaky_relu, slope 0.2
        ((float*)&st)[hh] = v;
        atomicMax(&emaxu[d * 4 + hh], f2o(v));
    }
    *(float4*)(ebuf + (size_t)e * 4) = st;
}

// ---------------------------------------------------------------------------
// 5) Decode max buffer in place (uint -> float; empty segment -> 0)
// ---------------------------------------------------------------------------
__global__ void k_decode(unsigned* __restrict__ emaxu, int n) {
    int t = blockIdx.x * blockDim.x + threadIdx.x;
    if (t >= n) return;
    unsigned u = emaxu[t];
    float f = (u == 0u) ? 0.0f : o2f(u);
    ((float*)emaxu)[t] = f;
}

// ---------------------------------------------------------------------------
// 6) Edge pass 2: ex = exp(e - emax[dst]); segment-sum denom via atomicAdd.
// ---------------------------------------------------------------------------
__global__ void k_edge2(const int* __restrict__ dst, const float* __restrict__ emaxf,
                        float* __restrict__ ebuf, float* __restrict__ denom, int E) {
    int t = blockIdx.x * blockDim.x + threadIdx.x;
    if (t >= E * HEADS) return;
    int e = t >> 2, hh = t & 3;
    int d = dst[e];
    float ex = __expf(ebuf[t] - emaxf[d * 4 + hh]);
    ebuf[t] = ex;
    atomicAdd(&denom[d * 4 + hh], ex);
}

// ---------------------------------------------------------------------------
// 7) Edge pass 3: out[dst] += alpha * h[src].  64 threads per edge (one per d),
//    each does 4 atomics (one per head). h and out live in L2.
// ---------------------------------------------------------------------------
__global__ void k_edge3(const int* __restrict__ src, const int* __restrict__ dst,
                        const float* __restrict__ ebuf, const float* __restrict__ denom,
                        const float* __restrict__ hfeat, float* __restrict__ out,
                        long long total) {
    long long t = (long long)blockIdx.x * blockDim.x + threadIdx.x;
    if (t >= total) return;
    int e = (int)(t >> 6);
    int j = (int)(t & 63);
    int s = src[e], d = dst[e];
    float4 ex = *(const float4*)(ebuf + (size_t)e * 4);
    float4 dn = *(const float4*)(denom + (size_t)d * 4);
    const float* hr = hfeat + (size_t)s * HD;
    float*       orr = out + (size_t)d * HD;
    float a0 = ex.x / (dn.x + 1e-9f);
    float a1 = ex.y / (dn.y + 1e-9f);
    float a2 = ex.z / (dn.z + 1e-9f);
    float a3 = ex.w / (dn.w + 1e-9f);
    atomicAdd(orr + 0 * DOUT + j, a0 * hr[0 * DOUT + j]);
    atomicAdd(orr + 1 * DOUT + j, a1 * hr[1 * DOUT + j]);
    atomicAdd(orr + 2 * DOUT + j, a2 * hr[2 * DOUT + j]);
    atomicAdd(orr + 3 * DOUT + j, a3 * hr[3 * DOUT + j]);
}

// ---------------------------------------------------------------------------
extern "C" void kernel_launch(void* const* d_in, const int* in_sizes, int n_in,
                              void* d_out, int out_size, void* d_ws, size_t ws_size,
                              hipStream_t stream) {
    const float* node_feats = (const float*)d_in[0];
    const float* edge_feats = (const float*)d_in[1];
    const float* W          = (const float*)d_in[2];
    const float* We         = (const float*)d_in[3];
    const float* attn_l     = (const float*)d_in[4];
    const float* attn_r     = (const float*)d_in[5];
    const float* attn_e     = (const float*)d_in[6];
    const int*   src        = (const int*)d_in[7];
    const int*   dst        = (const int*)d_in[8];

    const int N = in_sizes[0] / NODE_IN;
    const int E = in_sizes[7];
    float* out = (float*)d_out;

    // workspace carve-up (floats)
    float*    hfeat = (float*)d_ws;                       // N*256
    float*    ebuf  = hfeat + (size_t)N * HD;             // E*4
    float*    el    = ebuf + (size_t)E * HEADS;           // N*4
    float*    er    = el + (size_t)N * HEADS;             // N*4
    unsigned* emaxu = (unsigned*)(er + (size_t)N * HEADS);// N*4 (reused as float)
    float*    denom = (float*)(emaxu + (size_t)N * HEADS);// N*4
    float*    wered = denom + (size_t)N * HEADS;          // 64*4

    const int nh = N * HEADS;

    k_init<<<1024, 256, 0, stream>>>(out, denom, emaxu, N * HD, nh);
    k_wered<<<1, 256, 0, stream>>>(We, attn_e, wered);
    k_gemm<<<(N + 31) / 32, 256, 0, stream>>>(node_feats, W, hfeat, N);
    k_lr<<<(nh + 255) / 256, 256, 0, stream>>>(hfeat, attn_l, attn_r, el, er, N);
    k_edge1<<<(E + 255) / 256, 256, 0, stream>>>(edge_feats, wered, el, er, src, dst,
                                                 ebuf, emaxu, E);
    k_decode<<<(nh + 255) / 256, 256, 0, stream>>>(emaxu, nh);
    k_edge2<<<(E * HEADS + 255) / 256, 256, 0, stream>>>(dst, (const float*)emaxu,
                                                         ebuf, denom, E);
    long long total3 = (long long)E * DOUT;
    k_edge3<<<(unsigned)((total3 + 255) / 256), 256, 0, stream>>>(
        src, dst, ebuf, denom, hfeat, out, total3);
}
